// Temporal_GCN_27101243638196
// MI455X (gfx1250) — matrix operand order repaired and checked by reading, not compiled, not run on hardware
//
#include <hip/hip_runtime.h>
#include <hip/hip_bf16.h>

// ---------------------------------------------------------------------------
// Temporal GCN pipeline for MI455X (gfx1250, wave32, WMMA + TDM).
//   SpMM(native f32 atomic scatter, L2-resident gather)
//   -> bf16 WMMA GEMM, double-buffered TDM (tensor_load_to_lds) pipeline
//   -> BN -> skip -> ReLU (x2) -> pair gather
//   -> WMMA GEMM (+bias+ReLU, bf16 out) -> wave-per-row 256->2 projection.
// ---------------------------------------------------------------------------

typedef __bf16 bf16;
typedef __attribute__((ext_vector_type(16))) __bf16 v16bf;
typedef __attribute__((ext_vector_type(8)))  float  v8f;
typedef __attribute__((ext_vector_type(4)))  unsigned int u32x4;
typedef __attribute__((ext_vector_type(8)))  int          i32x8;
typedef __attribute__((ext_vector_type(4)))  int          i32x4;

union Frag16 { uint4 u[2]; v16bf v; };

#define N_NODES 100000
#define N_EDGES 3200000
#define D_IN    256
#define D_F     128
#define N_PAIRS 200000
#define BN_EPS  1e-3f

// --------------------------------------------------------------- TDM issue
// 2-D tensor tile load: Memory[row-major, stride0] -> LDS (contiguous tile),
// bf16 elements (data_size=2B). OOB rows/cols read as zero per ISA.
// D# layout per CDNA5 ISA 8.3/8.4 (group0: count/lds/global/type,
// group1: data_size, tensor dims, tile dims, stride; groups 2/3: zero = 2D).
__device__ __forceinline__ void tdm_load_2d(unsigned lds_off, const void* gaddr,
                                            unsigned dim0, unsigned dim1,
                                            unsigned stride0,
                                            unsigned tile0, unsigned tile1) {
    unsigned long long ga = (unsigned long long)gaddr;
    u32x4 g0;
    g0[0] = 1u;                                           // count=1, user D#
    g0[1] = lds_off;                                      // lds_addr (bytes)
    g0[2] = (unsigned)(ga & 0xFFFFFFFFu);                 // global_addr[31:0]
    g0[3] = (unsigned)((ga >> 32) & 0x1FFFFFFu)           // global_addr[56:32]
            | (2u << 30);                                 // type=2 ("image")
    i32x8 g1;
    g1[0] = (int)(1u << 16);                              // data_size=1 -> 2B
    g1[1] = (int)((dim0 & 0xFFFFu) << 16);                // tensor_dim0 lo
    g1[2] = (int)((dim0 >> 16) | ((dim1 & 0xFFFFu) << 16));
    g1[3] = (int)((dim1 >> 16) | (tile0 << 16));          // tile_dim0
    g1[4] = (int)(tile1 & 0xFFFFu);                       // tile_dim1, dim2=0
    g1[5] = (int)stride0;                                 // dim0_stride lo32
    g1[6] = 0;                                            // stride hi / dim1_stride
    g1[7] = 0;
    i32x4 gz = {0, 0, 0, 0};
#if __clang_major__ >= 23
    i32x8 gz8 = {0, 0, 0, 0, 0, 0, 0, 0};
    __builtin_amdgcn_tensor_load_to_lds(g0, g1, gz, gz, gz8, 0);
#else
    __builtin_amdgcn_tensor_load_to_lds(g0, g1, gz, gz, 0);
#endif
}

// ---------------------------------------------------------------- utilities
__global__ __launch_bounds__(256) void zero_f32_kernel(float* __restrict__ p, long n) {
    long i = (long)blockIdx.x * 256 + threadIdx.x;
    if (i < n) p[i] = 0.0f;
}

__global__ __launch_bounds__(256) void cvt_f32_bf16_kernel(const float* __restrict__ src,
                                                           bf16* __restrict__ dst, long n) {
    long i = (long)blockIdx.x * 256 + threadIdx.x;
    if (i < n) dst[i] = (bf16)src[i];
}

// ------------------------------------------------------------------- SpMM
// out[r,:] += v * x[c,:]   one wave (32 lanes) per edge, D/32 floats per lane
__global__ __launch_bounds__(256) void spmm_kernel(const int* __restrict__ rows,
                                                   const int* __restrict__ cols,
                                                   const float* __restrict__ vals,
                                                   const float* __restrict__ x,
                                                   float* __restrict__ out,
                                                   int E, int D) {
    int lane = threadIdx.x & 31;
    int e = blockIdx.x * 8 + (threadIdx.x >> 5);
    if (e >= E) return;
    int r = rows[e], c = cols[e];
    float v = vals[e];
    const float* xp = x + (long)c * D;
    float* op = out + (long)r * D;
    for (int j = lane; j < D; j += 32)
        unsafeAtomicAdd(op + j, v * xp[j]);   // native global_atomic_add_f32
}

// ------------------------------------------------------------ WMMA GEMM
// C[M,N] = A[M,K] * B[K,N], A/B bf16 row-major, f32 accumulate.
// Block: 256 threads = 8 waves, tile 128(M) x 64(N); wave = 2x2 wmma tiles.
// Double-buffered TDM pipeline: wave 0 issues stage s+1's descriptors into
// the alternate LDS buffer, then waits TENSORcnt<=2 (in-order completion =>
// stage s's transfers done) so stage s+1's DMA overlaps stage s's WMMAs.
template <bool RELU, bool BF16OUT>
__global__ __launch_bounds__(256)
void gemm_bf16_wmma(const bf16* __restrict__ A, const bf16* __restrict__ B,
                    const float* __restrict__ bias,
                    float* __restrict__ Cf, bf16* __restrict__ Cb,
                    int M, int N, int K) {
    __shared__ __align__(16) bf16 sA[2][128 * 32];   // 2 x 8 KB
    __shared__ __align__(16) bf16 sB[2][32 * 64];    // 2 x 4 KB

    const int t    = threadIdx.x;
    const int lane = t & 31;
    const int wave = t >> 5;
    const int wm   = wave >> 1;        // 0..3 -> 32-row slab
    const int wn   = wave & 1;         // 0..1 -> 32-col slab
    const int bm   = blockIdx.y * 128;
    const int bn   = blockIdx.x * 64;
    const int half = lane >> 4;        // 0: lanes 0-15, 1: lanes 16-31
    const int lrow = lane & 15;

    v8f acc[2][2] = {};

    const int nsteps = K >> 5;         // K / 32
    if (wave == 0) {                   // prologue: stage 0 into buffer 0
        tdm_load_2d((unsigned)(unsigned long long)(void*)sA[0],
                    A + (long)bm * K, (unsigned)K, (unsigned)(M - bm),
                    (unsigned)K, 32u, 128u);
        tdm_load_2d((unsigned)(unsigned long long)(void*)sB[0],
                    B + (long)bn, (unsigned)(N - bn), (unsigned)K,
                    (unsigned)N, 64u, 32u);
    }

    for (int s = 0; s < nsteps; ++s) {
        const int kk  = s << 5;
        const int cur = s & 1;
        if (wave == 0) {
            if (s + 1 < nsteps) {      // issue stage s+1 into other buffer
                tdm_load_2d((unsigned)(unsigned long long)(void*)sA[cur ^ 1],
                            A + (long)bm * K + kk + 32,
                            (unsigned)(K - kk - 32), (unsigned)(M - bm),
                            (unsigned)K, 32u, 128u);
                tdm_load_2d((unsigned)(unsigned long long)(void*)sB[cur ^ 1],
                            B + (long)(kk + 32) * N + bn,
                            (unsigned)(N - bn), (unsigned)(K - kk - 32),
                            (unsigned)N, 64u, 32u);
                __builtin_amdgcn_s_wait_tensorcnt(2);  // stage s complete
            } else {
                __builtin_amdgcn_s_wait_tensorcnt(0);
            }
        }
        __syncthreads();

        // ---- fragments ---------------------------------------------------
        // A (16x32 bf16): lane<16 holds row=lrow K{0..7,16..23};
        //                 lane>=16 holds K{8..15,24..31}
        Frag16 af[2];
        #pragma unroll
        for (int mt = 0; mt < 2; ++mt) {
            const bf16* ap = &sA[cur][(wm * 32 + mt * 16 + lrow) * 32 + half * 8];
            af[mt].u[0] = *(const uint4*)ap;
            af[mt].u[1] = *(const uint4*)(ap + 16);
        }
        // B (32x16 bf16): col=lrow per lane, K across vector elements in the
        // same order as A: lane<16 K{0..7,16..23}; lane>=16 K{8..15,24..31}.
        Frag16 bfr[2];
        #pragma unroll
        for (int nt = 0; nt < 2; ++nt) {
            const bf16* bp = &sB[cur][wn * 32 + nt * 16 + lrow];
            #pragma unroll
            for (int s = 0; s < 16; ++s)
                bfr[nt].v[s] = bp[((s & 7) + half * 8 + (s >> 3) * 16) * 64];
        }

        #pragma unroll
        for (int mt = 0; mt < 2; ++mt)
            #pragma unroll
            for (int nt = 0; nt < 2; ++nt)
                acc[mt][nt] = __builtin_amdgcn_wmma_f32_16x16x32_bf16(
                    false, af[mt].v, false, bfr[nt].v,
                    (short)0, acc[mt][nt], false, false);
        __syncthreads();
    }

    // ---- epilogue: C layout VGPR v -> row = v + 8*half, col = lrow -------
    #pragma unroll
    for (int mt = 0; mt < 2; ++mt) {
        #pragma unroll
        for (int nt = 0; nt < 2; ++nt) {
            int col = bn + wn * 32 + nt * 16 + lrow;
            float bv = bias ? bias[col] : 0.0f;
            #pragma unroll
            for (int v = 0; v < 8; ++v) {
                int row = bm + wm * 32 + mt * 16 + half * 8 + v;
                if (row < M) {
                    float y = acc[mt][nt][v] + bv;
                    if (RELU) y = fmaxf(y, 0.0f);
                    if (BF16OUT) Cb[(long)row * N + col] = (bf16)y;
                    else         Cf[(long)row * N + col] = y;
                }
            }
        }
    }
}

// ------------------------------------------------------------ BatchNorm
__global__ __launch_bounds__(256)
void bn_stats_kernel(const float* __restrict__ h, float* __restrict__ sum,
                     float* __restrict__ sumsq, int Nrows, int D) {
    int t = threadIdx.x;
    int c = t & (D - 1);
    int r = blockIdx.x * 256 + t / D;
    int rend = blockIdx.x * 256 + 256;
    if (rend > Nrows) rend = Nrows;
    int rstep = 256 / D;
    float s = 0.0f, s2 = 0.0f;
    for (; r < rend; r += rstep) {
        float v = h[(long)r * D + c];
        s += v; s2 += v * v;
    }
    atomicAdd(&sum[c], s);
    atomicAdd(&sumsq[c], s2);
}

__global__ void bn_finalize_kernel(const float* __restrict__ sum,
                                   const float* __restrict__ sumsq,
                                   const float* __restrict__ gamma,
                                   const float* __restrict__ beta,
                                   float* __restrict__ scale,
                                   float* __restrict__ shift, int Nrows, int D) {
    int c = threadIdx.x;
    if (c >= D) return;
    float inv_n = 1.0f / (float)Nrows;
    float mean  = sum[c] * inv_n;
    float var   = sumsq[c] * inv_n - mean * mean;
    float sc    = gamma[c] * rsqrtf(var + BN_EPS);
    scale[c] = sc;
    shift[c] = beta[c] - mean * sc;
}

// y = relu(h*scale + shift + skip), in place
__global__ __launch_bounds__(256)
void bn_apply_kernel(float* __restrict__ h, const float* __restrict__ scale,
                     const float* __restrict__ shift,
                     const float* __restrict__ skip, long total, int D) {
    long i = (long)blockIdx.x * 256 + threadIdx.x;
    if (i >= total) return;
    int c = (int)(i & (long)(D - 1));
    float y = h[i] * scale[c] + shift[c] + skip[i];
    h[i] = fmaxf(y, 0.0f);
}

// -------------------------------------------------- pair gather (f32->bf16)
__global__ __launch_bounds__(256)
void gather_pairs_kernel(const float* __restrict__ h2,
                         const long long* __restrict__ idx,
                         bf16* __restrict__ e, int Mrows) {
    long g = (long)blockIdx.x * 256 + threadIdx.x;   // over Mrows*256
    if (g >= (long)Mrows * 256) return;
    int m = (int)(g >> 8);
    int j = (int)(g & 255);
    int which = j >> 7;
    int col   = j & 127;
    long node = (long)idx[(long)m * 2 + which];
    e[g] = (bf16)h2[node * D_F + col];
}

// ------------------------------------ final 256 -> 2 projection, wave32 row
__global__ __launch_bounds__(256)
void classifier_out_kernel(const bf16* __restrict__ hid,
                           const float* __restrict__ Wc2,
                           const float* __restrict__ bc2,
                           float* __restrict__ out, int Mrows) {
    int lane = threadIdx.x & 31;
    int m = blockIdx.x * 8 + (threadIdx.x >> 5);
    if (m >= Mrows) return;
    const bf16* hp = hid + (long)m * 256;
    float a0 = 0.0f, a1 = 0.0f;
    #pragma unroll
    for (int k = 0; k < 8; ++k) {
        int j = lane + k * 32;
        float v = (float)hp[j];
        a0 += v * Wc2[j * 2 + 0];
        a1 += v * Wc2[j * 2 + 1];
    }
    for (int off = 16; off; off >>= 1) {
        a0 += __shfl_xor(a0, off, 32);
        a1 += __shfl_xor(a1, off, 32);
    }
    if (lane == 0) {
        out[(long)m * 2 + 0] = a0 + bc2[0];
        out[(long)m * 2 + 1] = a1 + bc2[1];
    }
}

// ---------------------------------------------------------------------------
extern "C" void kernel_launch(void* const* d_in, const int* in_sizes, int n_in,
                              void* d_out, int out_size, void* d_ws, size_t ws_size,
                              hipStream_t stream) {
    const int*       rows  = (const int*)d_in[0];
    const int*       cols  = (const int*)d_in[1];
    const float*     vals  = (const float*)d_in[2];
    const float*     feat  = (const float*)d_in[3];
    const long long* lidx  = (const long long*)d_in[4];
    const float*     W1    = (const float*)d_in[5];
    const float*     gamma1= (const float*)d_in[6];
    const float*     beta1 = (const float*)d_in[7];
    const float*     W2    = (const float*)d_in[8];
    const float*     gamma2= (const float*)d_in[9];
    const float*     beta2 = (const float*)d_in[10];
    const float*     skip2 = (const float*)d_in[11];
    const float*     Wc1   = (const float*)d_in[12];
    const float*     bc1   = (const float*)d_in[13];
    const float*     Wc2   = (const float*)d_in[14];
    const float*     bc2   = (const float*)d_in[15];
    float* out = (float*)d_out;

    const int N = N_NODES, E = N_EDGES, D = D_IN, DF = D_F, MP = N_PAIRS;

    // ---- workspace carve-out -------------------------------------------
    char* ws = (char*)d_ws;
    size_t off = 0;
    auto carve = [&](size_t bytes) -> char* {
        char* p = ws + off;
        off += (bytes + 255) & ~(size_t)255;
        return p;
    };
    float* agg   = (float*)carve((size_t)N * D * 4);        // SpMM accumulator
    bf16*  actb  = (bf16*) carve((size_t)N * D * 2);        // bf16 activations
    float* h1    = (float*)carve((size_t)N * D * 4);        // layer1 output
    float* h2    = (float*)carve((size_t)N * DF * 4);       // layer2 output
    float* skipf = (float*)carve((size_t)N * DF * 4);       // skip path
    bf16*  tmpb  = (bf16*) carve((size_t)N * D * 2);        // bf16 of agg2
    bf16*  eb    = (bf16*) carve((size_t)MP * 256 * 2);     // gathered pairs
    bf16*  hidb  = (bf16*) carve((size_t)MP * 256 * 2);     // classifier hidden
    bf16*  w1b   = (bf16*) carve((size_t)256 * 256 * 2);
    bf16*  w2b   = (bf16*) carve((size_t)256 * 128 * 2);
    bf16*  skpb  = (bf16*) carve((size_t)256 * 128 * 2);
    bf16*  wc1b  = (bf16*) carve((size_t)256 * 256 * 2);
    float* stats = (float*)carve((size_t)4 * 256 * 4);
    float* s_sum = stats, *s_sq = stats + 256, *s_sc = stats + 512, *s_sh = stats + 768;

    auto blks = [](long n) { return (unsigned)((n + 255) / 256); };

    // ---- weight conversions (bf16) -------------------------------------
    cvt_f32_bf16_kernel<<<blks(256*256), 256, 0, stream>>>(W1,    w1b,  256*256);
    cvt_f32_bf16_kernel<<<blks(256*128), 256, 0, stream>>>(W2,    w2b,  256*128);
    cvt_f32_bf16_kernel<<<blks(256*128), 256, 0, stream>>>(skip2, skpb, 256*128);
    cvt_f32_bf16_kernel<<<blks(256*256), 256, 0, stream>>>(Wc1,   wc1b, 256*256);

    // ---- layer 1 ---------------------------------------------------------
    zero_f32_kernel<<<blks((long)N*D), 256, 0, stream>>>(agg, (long)N*D);
    spmm_kernel<<<(E+7)/8, 256, 0, stream>>>(rows, cols, vals, feat, agg, E, D);
    cvt_f32_bf16_kernel<<<blks((long)N*D), 256, 0, stream>>>(agg, actb, (long)N*D);
    gemm_bf16_wmma<false,false><<<dim3(D/64, (N+127)/128), 256, 0, stream>>>(
        actb, w1b, nullptr, h1, nullptr, N, D, D);
    zero_f32_kernel<<<2, 256, 0, stream>>>(stats, 512);
    bn_stats_kernel<<<(N+255)/256, 256, 0, stream>>>(h1, s_sum, s_sq, N, D);
    bn_finalize_kernel<<<1, 256, 0, stream>>>(s_sum, s_sq, gamma1, beta1, s_sc, s_sh, N, D);
    bn_apply_kernel<<<blks((long)N*D), 256, 0, stream>>>(h1, s_sc, s_sh, feat, (long)N*D, D);

    // ---- layer 2 ---------------------------------------------------------
    cvt_f32_bf16_kernel<<<blks((long)N*D), 256, 0, stream>>>(h1, actb, (long)N*D);
    zero_f32_kernel<<<blks((long)N*D), 256, 0, stream>>>(agg, (long)N*D);
    spmm_kernel<<<(E+7)/8, 256, 0, stream>>>(rows, cols, vals, h1, agg, E, D);
    gemm_bf16_wmma<false,false><<<dim3(DF/64, (N+127)/128), 256, 0, stream>>>(
        actb, skpb, nullptr, skipf, nullptr, N, DF, D);
    cvt_f32_bf16_kernel<<<blks((long)N*D), 256, 0, stream>>>(agg, tmpb, (long)N*D);
    gemm_bf16_wmma<false,false><<<dim3(DF/64, (N+127)/128), 256, 0, stream>>>(
        tmpb, w2b, nullptr, h2, nullptr, N, DF, D);
    zero_f32_kernel<<<2, 256, 0, stream>>>(stats, 512);
    bn_stats_kernel<<<(N+255)/256, 256, 0, stream>>>(h2, s_sum, s_sq, N, DF);
    bn_finalize_kernel<<<1, 128, 0, stream>>>(s_sum, s_sq, gamma2, beta2, s_sc, s_sh, N, DF);
    bn_apply_kernel<<<blks((long)N*DF), 256, 0, stream>>>(h2, s_sc, s_sh, skipf, (long)N*DF, DF);

    // ---- classifier ------------------------------------------------------
    gather_pairs_kernel<<<blks((long)MP*256), 256, 0, stream>>>(h2, lidx, eb, MP);
    gemm_bf16_wmma<true,true><<<dim3(256/64, (MP+127)/128), 256, 0, stream>>>(
        eb, wc1b, bc1, nullptr, hidb, MP, 256, 256);
    classifier_out_kernel<<<(MP+7)/8, 256, 0, stream>>>(hidb, Wc2, bc2, out, MP);
}
